// CombinedNGPNeRFW_12841952215766
// MI455X (gfx1250) — matrix operand order repaired and checked by reading, not compiled, not run on hardware
//
#include <hip/hip_runtime.h>

typedef _Float16 f16x;
typedef __attribute__((ext_vector_type(16))) _Float16 v16h;
typedef __attribute__((ext_vector_type(8)))  float    v8f;

union FragU { v16h h; uint4 u[2]; };

__constant__ float NLc[16] = {16.f,22.f,30.f,42.f,58.f,80.f,111.f,154.f,
                              213.f,295.f,407.f,563.f,777.f,1074.f,1483.f,2048.f};

#define HASH_T 524288u
#define HASH_MASK (HASH_T - 1u)

// ---------------- WMMA helpers ----------------

__device__ inline v8f wmma_f16(v16h a, v16h b, v8f c) {
  return __builtin_amdgcn_wmma_f32_16x16x32_f16(false, a, false, b, (short)0, c, false, false);
}

// A-fragment (16x32 f16) from a row-major LDS tile [16][stride].
// ISA layout: lane L<16: M=L, elems 0..7 = K 0..7, elems 8..15 = K 16..23
//             lane L>=16: M=L-16, elems 0..7 = K 8..15, elems 8..15 = K 24..31
__device__ inline v16h loadA_lds(const f16x* tile, int stride, int ktile) {
  int lane = threadIdx.x & 31;
  int row  = lane & 15;
  int kh   = lane >> 4;
  const f16x* p = tile + row * stride + ktile * 32 + kh * 8;
  FragU f;
  f.u[0] = *(const uint4*)(p);
  f.u[1] = *(const uint4*)(p + 16);
  return f.h;
}

// B-fragment pre-swizzled in global workspace: frag*512 + lane*16 f16.
__device__ inline v16h loadB_g(const f16x* wz, int frag) {
  int lane = threadIdx.x & 31;
  const f16x* p = wz + (size_t)frag * 512 + lane * 16;
  FragU f;
  f.u[0] = *(const uint4*)(p);
  f.u[1] = *(const uint4*)(p + 8);
  return f.h;
}

__device__ inline void wsync() { __builtin_amdgcn_wave_barrier(); }

// D-fragment -> LDS f16 tile with bias + ReLU.
// D layout: VGPR v, lane L<16: M=v,N=L ; lane L>=16: M=v+8,N=L-16
__device__ inline void storeD_relu(f16x* tile, int stride, int nt, const v8f& c,
                                   const float* bias) {
  int lane = threadIdx.x & 31;
  int col  = (lane & 15) + nt * 16;
  float b  = bias[col];
  int rb   = (lane < 16) ? 0 : 8;
#pragma unroll
  for (int v2 = 0; v2 < 8; ++v2) {
    float val = c[v2] + b;
    val = val > 0.f ? val : 0.f;
    tile[(rb + v2) * stride + col] = (f16x)val;
  }
}

__device__ inline float softplusf(float x) {
  return x > 20.f ? x : log1pf(expf(x));
}

// ---------------- weight prep: f32 -> f16 B-fragments ----------------
// frag map: 0-3 Wd1(32x64, scaled 1/256) | 4-5 Wd2(64x16) | 6-17 Ws1(91x64 pad96)
// 18-25 Ws2(64x64) | 26-27 Ws3(64x3 pad16) | 28-31 Wt1(32x64) | 32-39 Wt2 | 40-47 Wt3
// 48-49 heads(64x5 pad16: col0=Wtd, col1-3=Wtr, col4=Wtb)

__global__ void nerfw_prep(const float* __restrict__ Wd1, const float* __restrict__ Wd2,
                           const float* __restrict__ Ws1, const float* __restrict__ Ws2,
                           const float* __restrict__ Ws3, const float* __restrict__ Wt1,
                           const float* __restrict__ Wt2, const float* __restrict__ Wt3,
                           const float* __restrict__ Wtd, const float* __restrict__ Wtr,
                           const float* __restrict__ Wtb, f16x* __restrict__ wz) {
  int e = blockIdx.x * blockDim.x + threadIdx.x;
  if (e >= 50 * 512) return;
  int frag = e >> 9;
  int L    = (e >> 4) & 31;
  int j    = e & 15;
  int klo  = ((j >> 3) << 4) + ((L >> 4) << 3) + (j & 7);   // K within a 32-tile
  int n16  = L & 15;                // column within a 16-tile
  float v = 0.f;
  if (frag < 4) {
    int k = klo, n = frag * 16 + n16;
    v = Wd1[k * 64 + n] * (1.f / 256.f);
  } else if (frag < 6) {
    int k = (frag - 4) * 32 + klo;
    v = Wd2[k * 16 + n16];
  } else if (frag < 18) {
    int f = frag - 6; int kt = f >> 2; int nt = f & 3;
    int k = kt * 32 + klo, n = nt * 16 + n16;
    v = (k < 91) ? Ws1[k * 64 + n] : 0.f;
  } else if (frag < 26) {
    int f = frag - 18; int kt = f >> 2; int nt = f & 3;
    int k = kt * 32 + klo, n = nt * 16 + n16;
    v = Ws2[k * 64 + n];
  } else if (frag < 28) {
    int k = (frag - 26) * 32 + klo;
    v = (n16 < 3) ? Ws3[k * 3 + n16] : 0.f;
  } else if (frag < 32) {
    int k = klo, n = (frag - 28) * 16 + n16;
    v = Wt1[k * 64 + n];
  } else if (frag < 40) {
    int f = frag - 32; int kt = f >> 2; int nt = f & 3;
    int k = kt * 32 + klo, n = nt * 16 + n16;
    v = Wt2[k * 64 + n];
  } else if (frag < 48) {
    int f = frag - 40; int kt = f >> 2; int nt = f & 3;
    int k = kt * 32 + klo, n = nt * 16 + n16;
    v = Wt3[k * 64 + n];
  } else {
    int k = (frag - 48) * 32 + klo;
    if (n16 == 0)      v = Wtd[k];
    else if (n16 < 4)  v = Wtr[k * 3 + (n16 - 1)];
    else if (n16 == 4) v = Wtb[k];
  }
  wz[(size_t)frag * 512 + L * 16 + j] = (f16x)v;
}

// ---------------- fused forward: one wave = one 16-point tile ----------------

__global__ __launch_bounds__(256) void nerfw_main(
    const float* __restrict__ xg, const float* __restrict__ dg,
    const int* __restrict__ aidx, const int* __restrict__ tidx,
    const float* __restrict__ tables,
    const float* __restrict__ emb_a, const float* __restrict__ emb_t,
    const float* __restrict__ bd1, const float* __restrict__ bd2,
    const float* __restrict__ bs1, const float* __restrict__ bs2,
    const float* __restrict__ bs3, const float* __restrict__ bt1,
    const float* __restrict__ bt2, const float* __restrict__ bt3,
    const float* __restrict__ btd, const float* __restrict__ btr,
    const float* __restrict__ btb,
    const f16x* __restrict__ wz, float* __restrict__ out, int N) {
  __shared__ alignas(16) f16x sInS[8 * 16 * 96];  // [h(16) | denc(27) | a_emb(48) | 0(5)]
  __shared__ alignas(16) f16x tInS[8 * 16 * 40];  // [h(16) | t_emb(16)] stride 40
  __shared__ alignas(16) f16x tBS [8 * 16 * 72];  // feats / hidden acts, stride 72
  __shared__ float xnS[8 * 16 * 4];
  __shared__ float mkS[8 * 16];
  __shared__ float h0S[8 * 16];

  int wave = threadIdx.x >> 5;
  int lane = threadIdx.x & 31;
  int tiles = (N + 15) >> 4;
  int tile = blockIdx.x * 8 + wave;
  if (tile >= tiles) return;          // wave-uniform
  int base = tile * 16;

  f16x* sIn = sInS + wave * 16 * 96;
  f16x* tIn = tInS + wave * 16 * 40;
  f16x* tB  = tBS  + wave * 16 * 72;
  float* xn = xnS + wave * 64;
  float* mk = mkS + wave * 16;
  float* h0 = h0S + wave * 16;

  // ---- Phase 1: per-point scalars, pos-enc, embedding gathers ----
  if (lane < 16) {
    int p = lane, g = base + p;
    float x0 = xg[g * 3 + 0], x1 = xg[g * 3 + 1], x2 = xg[g * 3 + 2];
    float n0 = x0 * 0.25f, n1 = x1 * 0.25f, n2 = x2 * 0.25f;
    bool m = (fabsf(n0) < 0.5f) & (fabsf(n1) < 0.5f) & (fabsf(n2) < 0.5f);
    mk[p] = m ? 1.f : 0.f;
    xn[p * 4 + 0] = n0 + 0.5f;
    xn[p * 4 + 1] = n1 + 0.5f;
    xn[p * 4 + 2] = n2 + 0.5f;
    float d0 = dg[g * 3 + 0], d1 = dg[g * 3 + 1], d2 = dg[g * 3 + 2];
    sIn[p * 96 + 16] = (f16x)d0;
    sIn[p * 96 + 17] = (f16x)d1;
    sIn[p * 96 + 18] = (f16x)d2;
    float sc = 1.f;
#pragma unroll
    for (int jj = 0; jj < 4; ++jj) {
      int cb = 19 + jj * 6;
      sIn[p * 96 + cb + 0] = (f16x)sinf(sc * d0);
      sIn[p * 96 + cb + 1] = (f16x)sinf(sc * d1);
      sIn[p * 96 + cb + 2] = (f16x)sinf(sc * d2);
      sIn[p * 96 + cb + 3] = (f16x)cosf(sc * d0);
      sIn[p * 96 + cb + 4] = (f16x)cosf(sc * d1);
      sIn[p * 96 + cb + 5] = (f16x)cosf(sc * d2);
      sc *= 2.f;
    }
#pragma unroll
    for (int cc = 91; cc < 96; ++cc) sIn[p * 96 + cc] = (f16x)0.f;
  }
  {
    int p = lane & 15, half = lane >> 4, g = base + p;
    int ai = aidx[g];
#pragma unroll
    for (int i = 0; i < 24; ++i)
      sIn[p * 96 + 43 + half * 24 + i] = (f16x)emb_a[(size_t)ai * 48 + half * 24 + i];
    int ti = tidx[g];
#pragma unroll
    for (int i = 0; i < 8; ++i)
      tIn[p * 40 + 16 + half * 8 + i] = (f16x)emb_t[(size_t)ti * 16 + half * 8 + i];
  }
  wsync();

  // ---- Phase 2: hash-grid features (8 point-level pairs per lane) ----
#pragma unroll
  for (int i = 0; i < 8; ++i) {
    int pl = i * 32 + lane;
    int p = pl & 15, lvl = pl >> 4;
    float nx = xn[p * 4 + 0], ny = xn[p * 4 + 1], nz = xn[p * 4 + 2];
    float Nl = NLc[lvl];
    float xs0 = nx * Nl, xs1 = ny * Nl, xs2 = nz * Nl;
    float f0 = floorf(xs0), f1 = floorf(xs1), f2 = floorf(xs2);
    float l0 = xs0 - f0, l1 = xs1 - f1, l2 = xs2 - f2;
    unsigned i0 = (unsigned)(int)f0, i1 = (unsigned)(int)f1, i2 = (unsigned)(int)f2;
    float a0 = 0.f, a1 = 0.f;
#pragma unroll
    for (int c = 0; c < 8; ++c) {
      unsigned ox = c & 1u, oy = (c >> 1) & 1u, oz = (c >> 2) & 1u;
      unsigned h = ((i0 + ox) * 1u) ^ ((i1 + oy) * 2654435761u) ^ ((i2 + oz) * 805459861u);
      h &= HASH_MASK;
      const float2 f = *(const float2*)(tables + ((size_t)lvl * HASH_T + h) * 2);
      float w = (ox ? l0 : 1.f - l0) * (oy ? l1 : 1.f - l1) * (oz ? l2 : 1.f - l2);
      a0 += w * f.x;
      a1 += w * f.y;
    }
    // x256 to keep tiny features out of f16 subnormal range (Wd1 pre-scaled 1/256)
    tB[p * 72 + 2 * lvl + 0] = (f16x)(a0 * 256.f);
    tB[p * 72 + 2 * lvl + 1] = (f16x)(a1 * 256.f);
  }
  wsync();

  v8f cz = {0.f, 0.f, 0.f, 0.f, 0.f, 0.f, 0.f, 0.f};
  v8f acc[4];

  // ---- L1: feats(16x32) @ Wd1 -> 16x64, ReLU ----
  {
    v16h aF = loadA_lds(tB, 72, 0);
#pragma unroll
    for (int nt = 0; nt < 4; ++nt)
      acc[nt] = wmma_f16(aF, loadB_g(wz, nt), cz);
    wsync();
#pragma unroll
    for (int nt = 0; nt < 4; ++nt) storeD_relu(tB, 72, nt, acc[nt], bd1);
    wsync();
  }

  // ---- L2: 16x64 @ Wd2 -> h 16x16 (no activation) ----
  {
    v16h a0 = loadA_lds(tB, 72, 0);
    v16h a1 = loadA_lds(tB, 72, 1);
    v8f hc = wmma_f16(a0, loadB_g(wz, 4), cz);
    hc = wmma_f16(a1, loadB_g(wz, 5), hc);
    int col = lane & 15;
    float b = bd2[col];
    int rb = (lane < 16) ? 0 : 8;
#pragma unroll
    for (int v2 = 0; v2 < 8; ++v2) {
      float val = hc[v2] + b;
      int m = rb + v2;
      sIn[m * 96 + col] = (f16x)val;
      tIn[m * 40 + col] = (f16x)val;
      if (col == 0) h0[m] = val;
    }
    wsync();
  }

  // ---- Static branch: sIn(16x96) @ Ws1 ReLU @ Ws2 ReLU @ Ws3 sigmoid ----
  {
    v16h a0 = loadA_lds(sIn, 96, 0);
    v16h a1 = loadA_lds(sIn, 96, 1);
    v16h a2 = loadA_lds(sIn, 96, 2);
#pragma unroll
    for (int nt = 0; nt < 4; ++nt) {
      v8f t = wmma_f16(a0, loadB_g(wz, 6 + nt), cz);
      t = wmma_f16(a1, loadB_g(wz, 10 + nt), t);
      t = wmma_f16(a2, loadB_g(wz, 14 + nt), t);
      acc[nt] = t;
    }
    wsync();
#pragma unroll
    for (int nt = 0; nt < 4; ++nt) storeD_relu(tB, 72, nt, acc[nt], bs1);
    wsync();
  }
  {
    v16h a0 = loadA_lds(tB, 72, 0);
    v16h a1 = loadA_lds(tB, 72, 1);
#pragma unroll
    for (int nt = 0; nt < 4; ++nt) {
      v8f t = wmma_f16(a0, loadB_g(wz, 18 + nt), cz);
      t = wmma_f16(a1, loadB_g(wz, 22 + nt), t);
      acc[nt] = t;
    }
    wsync();
#pragma unroll
    for (int nt = 0; nt < 4; ++nt) storeD_relu(tB, 72, nt, acc[nt], bs2);
    wsync();
  }
  {
    v16h a0 = loadA_lds(tB, 72, 0);
    v16h a1 = loadA_lds(tB, 72, 1);
    v8f s3 = wmma_f16(a0, loadB_g(wz, 26), cz);
    s3 = wmma_f16(a1, loadB_g(wz, 27), s3);
    int n = lane & 15;
    int rb = (lane < 16) ? 0 : 8;
    if (n < 3) {
      float b = bs3[n];
#pragma unroll
      for (int v2 = 0; v2 < 8; ++v2) {
        int m = rb + v2;
        float val = 1.f / (1.f + expf(-(s3[v2] + b)));
        out[(size_t)(base + m) * 3 + n] = (mk[m] != 0.f) ? val : 0.f;
      }
    }
    if (lane < 16) {  // static_sigma = exp(h[:,0]) masked
      out[(size_t)3 * N + base + lane] = (mk[lane] != 0.f) ? expf(h0[lane]) : 0.f;
    }
  }

  // ---- Transient branch: tIn(16x32) @ Wt1 ReLU @ Wt2 ReLU @ Wt3 ReLU @ heads ----
  {
    v16h aT = loadA_lds(tIn, 40, 0);
#pragma unroll
    for (int nt = 0; nt < 4; ++nt)
      acc[nt] = wmma_f16(aT, loadB_g(wz, 28 + nt), cz);
    wsync();
#pragma unroll
    for (int nt = 0; nt < 4; ++nt) storeD_relu(tB, 72, nt, acc[nt], bt1);
    wsync();
  }
  {
    v16h a0 = loadA_lds(tB, 72, 0);
    v16h a1 = loadA_lds(tB, 72, 1);
#pragma unroll
    for (int nt = 0; nt < 4; ++nt) {
      v8f t = wmma_f16(a0, loadB_g(wz, 32 + nt), cz);
      t = wmma_f16(a1, loadB_g(wz, 36 + nt), t);
      acc[nt] = t;
    }
    wsync();
#pragma unroll
    for (int nt = 0; nt < 4; ++nt) storeD_relu(tB, 72, nt, acc[nt], bt2);
    wsync();
  }
  {
    v16h a0 = loadA_lds(tB, 72, 0);
    v16h a1 = loadA_lds(tB, 72, 1);
#pragma unroll
    for (int nt = 0; nt < 4; ++nt) {
      v8f t = wmma_f16(a0, loadB_g(wz, 40 + nt), cz);
      t = wmma_f16(a1, loadB_g(wz, 44 + nt), t);
      acc[nt] = t;
    }
    wsync();
#pragma unroll
    for (int nt = 0; nt < 4; ++nt) storeD_relu(tB, 72, nt, acc[nt], bt3);
    wsync();
  }
  {
    v16h a0 = loadA_lds(tB, 72, 0);
    v16h a1 = loadA_lds(tB, 72, 1);
    v8f hd = wmma_f16(a0, loadB_g(wz, 48), cz);
    hd = wmma_f16(a1, loadB_g(wz, 49), hd);
    int n = lane & 15;
    int rb = (lane < 16) ? 0 : 8;
    if (n == 0) {          // transient_sigma = exp(softplus(.)) masked
      float b = btd[0];
#pragma unroll
      for (int v2 = 0; v2 < 8; ++v2) {
        int m = rb + v2;
        float sp = softplusf(hd[v2] + b);
        out[(size_t)7 * N + base + m] = (mk[m] != 0.f) ? expf(sp) : 0.f;
      }
    } else if (n < 4) {    // transient_rgb = sigmoid masked
      float b = btr[n - 1];
#pragma unroll
      for (int v2 = 0; v2 < 8; ++v2) {
        int m = rb + v2;
        float val = 1.f / (1.f + expf(-(hd[v2] + b)));
        out[(size_t)4 * N + (size_t)(base + m) * 3 + (n - 1)] = (mk[m] != 0.f) ? val : 0.f;
      }
    } else if (n == 4) {   // transient_beta = softplus + 0.1, else 0.1
      float b = btb[0];
#pragma unroll
      for (int v2 = 0; v2 < 8; ++v2) {
        int m = rb + v2;
        float sp = softplusf(hd[v2] + b) + 0.1f;
        out[(size_t)8 * N + base + m] = (mk[m] != 0.f) ? sp : 0.1f;
      }
    }
  }
}

extern "C" void kernel_launch(void* const* d_in, const int* in_sizes, int n_in,
                              void* d_out, int out_size, void* d_ws, size_t ws_size,
                              hipStream_t stream) {
  const float* x      = (const float*)d_in[0];
  const float* d      = (const float*)d_in[1];
  const int*   aidx   = (const int*)d_in[2];
  const int*   tidx   = (const int*)d_in[3];
  const float* tables = (const float*)d_in[4];
  const float* emb_a  = (const float*)d_in[5];
  const float* emb_t  = (const float*)d_in[6];
  const float* Wd1 = (const float*)d_in[7];  const float* bd1 = (const float*)d_in[8];
  const float* Wd2 = (const float*)d_in[9];  const float* bd2 = (const float*)d_in[10];
  const float* Ws1 = (const float*)d_in[11]; const float* bs1 = (const float*)d_in[12];
  const float* Ws2 = (const float*)d_in[13]; const float* bs2 = (const float*)d_in[14];
  const float* Ws3 = (const float*)d_in[15]; const float* bs3 = (const float*)d_in[16];
  const float* Wt1 = (const float*)d_in[17]; const float* bt1 = (const float*)d_in[18];
  const float* Wt2 = (const float*)d_in[19]; const float* bt2 = (const float*)d_in[20];
  const float* Wt3 = (const float*)d_in[21]; const float* bt3 = (const float*)d_in[22];
  const float* Wtd = (const float*)d_in[23]; const float* btd = (const float*)d_in[24];
  const float* Wtr = (const float*)d_in[25]; const float* btr = (const float*)d_in[26];
  const float* Wtb = (const float*)d_in[27]; const float* btb = (const float*)d_in[28];
  float* out = (float*)d_out;
  int N = in_sizes[0] / 3;
  f16x* wz = (f16x*)d_ws;  // 50 frags * 1 KB = 51200 B

  hipLaunchKernelGGL(nerfw_prep, dim3((50 * 512 + 255) / 256), dim3(256), 0, stream,
                     Wd1, Wd2, Ws1, Ws2, Ws3, Wt1, Wt2, Wt3, Wtd, Wtr, Wtb, wz);

  int tiles  = (N + 15) / 16;
  int blocks = (tiles + 7) / 8;
  hipLaunchKernelGGL(nerfw_main, dim3(blocks), dim3(256), 0, stream,
                     x, d, aidx, tidx, tables, emb_a, emb_t,
                     bd1, bd2, bs1, bs2, bs3, bt1, bt2, bt3, btd, btr, btb,
                     wz, out, N);
}
